// LiLTMultiHeadSelfAttention_88587995448051
// MI455X (gfx1250) — matrix operand order repaired and checked by reading, not compiled, not run on hardware
//
#include <hip/hip_runtime.h>
#include <hip/hip_bf16.h>

// ---------------- shapes (fixed by the reference) ----------------
#define BATCH 8
#define SEQ   1024
#define HEADS 12
#define DMODEL 768
#define DHEAD 64
#define DLAY  192
#define DHL   16
#define DCAT  96            // 64 (q) + 16 (lq) + 16 zero pad -> 3 x K32 WMMA steps
#define NEGF  (-3.4028234663852886e+38f)

typedef __attribute__((ext_vector_type(16))) __bf16 v16bf;
typedef __attribute__((ext_vector_type(8)))  float  v8f;
typedef __attribute__((ext_vector_type(4)))  float  v4f;   // native vector for NT stores

union FragU { unsigned short s[16]; v16bf v; };

__device__ __forceinline__ unsigned short f2bf(float f) {
  unsigned int x = __builtin_bit_cast(unsigned int, f);
  x += 0x7FFFu + ((x >> 16) & 1u);      // round-to-nearest-even
  return (unsigned short)(x >> 16);
}

__device__ __forceinline__ void unpack8(uint4 q, unsigned short* s) {
  s[0] = (unsigned short)(q.x); s[1] = (unsigned short)(q.x >> 16);
  s[2] = (unsigned short)(q.y); s[3] = (unsigned short)(q.y >> 16);
  s[4] = (unsigned short)(q.z); s[5] = (unsigned short)(q.z >> 16);
  s[6] = (unsigned short)(q.w); s[7] = (unsigned short)(q.w >> 16);
}

__device__ __forceinline__ v8f wmma_bf16(const FragU& a, const FragU& b, v8f c) {
  return __builtin_amdgcn_wmma_f32_16x16x32_bf16(false, a.v, false, b.v,
                                                 (short)0, c, false, false);
}

// ---------------- fp32 -> bf16 conversion ----------------
__global__ void cvt_f32_bf16(const float* __restrict__ src,
                             unsigned short* __restrict__ dst, int n) {
  int i = blockIdx.x * blockDim.x + threadIdx.x;
  if (i < n) dst[i] = f2bf(src[i]);
}

// ---------------- projection GEMM: out = (A @ Bw^T + bias) * scale -------------
// A: [M,K] bf16 row-major. Bw: [N,K] bf16 row-major (torch weight layout).
// block = 128 threads = 4 waves; wave w owns the 16x16 tile at
// (blockIdx.x*16, blockIdx.y*64 + w*16).
// mode 0: out[m*N+n]
// mode 1: headT layout  [B, N, 1024]   (vT / lvT)
// mode 2: cat layout    [B, H, S, 96] at column (n%dh)+cbase  (qcat / kcat)
__global__ __launch_bounds__(128)
void gemm_proj_wmma(const unsigned short* __restrict__ A,
                    const unsigned short* __restrict__ Bw,
                    const float* __restrict__ bias,
                    unsigned short* __restrict__ out,
                    int N, int K, float scale, int mode, int dh, int cbase) {
  const int lane = threadIdx.x & 31;
  const int wave = threadIdx.x >> 5;
  const int mBase = blockIdx.x * 16;
  const int nBase = blockIdx.y * 64 + wave * 16;
  const int m    = lane & 15;
  const int half = lane >> 4;

  const unsigned short* arow = A  + (size_t)(mBase + m) * K + half * 8;
  const unsigned short* brow = Bw + (size_t)(nBase + m) * K + half * 8;

  v8f acc = {0.f, 0.f, 0.f, 0.f, 0.f, 0.f, 0.f, 0.f};
  for (int k = 0; k < K; k += 32) {
    FragU fa, fb;
    unpack8(*(const uint4*)(arow + k),      fa.s);
    unpack8(*(const uint4*)(arow + k + 16), fa.s + 8);
    unpack8(*(const uint4*)(brow + k),      fb.s);
    unpack8(*(const uint4*)(brow + k + 16), fb.s + 8);
    acc = wmma_bf16(fa, fb, acc);
  }

  const int n = nBase + m;
  const float bv = bias[n];
#pragma unroll
  for (int j = 0; j < 8; ++j) {
    const int row = mBase + half * 8 + j;          // D layout: m = 8*(lane>>4)+j
    const float val = (acc[j] + bv) * scale;
    size_t off;
    if (mode == 0) {
      off = (size_t)row * N + n;
    } else if (mode == 1) {                        // [B, N, 1024]
      off = (size_t)(row >> 10) * ((size_t)N << 10) + ((size_t)n << 10) + (row & 1023);
    } else {                                       // [B, H, S, 96]
      off = (size_t)(row >> 10) * (HEADS * SEQ * DCAT)
          + (size_t)(n / dh) * (SEQ * DCAT)
          + (size_t)(row & 1023) * DCAT
          + (n % dh) + cbase;
    }
    out[off] = f2bf(val);
  }
}

// ---------------- fused attention per (b, h, 32-row q tile) ----------------
// 256 threads = 8 waves. Dynamic LDS:
//   sc  : 32x1024 fp32 score tile       (128 KB)
//   red : 32x8 fp32 reduction scratch   (1 KB)
//   pb  : 32x1024 bf16 probs tile       (64 KB)   total 197632 B < 320 KB WGP LDS
__global__ __launch_bounds__(256)
void attn_wmma(const unsigned short* __restrict__ qcat,
               const unsigned short* __restrict__ kcat,
               const unsigned short* __restrict__ vT,
               const unsigned short* __restrict__ lvT,
               const float* __restrict__ mask,
               float* __restrict__ ctx, float* __restrict__ lctx,
               float* __restrict__ probs) {
  extern __shared__ float smem[];
  float* sc  = smem;                                   // [32][1024] fp32
  float* red = smem + 32 * 1024;                       // [32][8]
  unsigned short* pb = (unsigned short*)(red + 32 * 8); // [32][1024] bf16

  const int idx = blockIdx.x;
  const int qt  = idx & 31;                 // S/32 tiles
  const int h   = (idx >> 5) % HEADS;
  const int b   = idx / (32 * HEADS);
  const int tid = threadIdx.x;
  const int lane = tid & 31;
  const int wave = tid >> 5;
  const size_t bh = (size_t)(b * HEADS + h);

  const unsigned short* qbase = qcat + bh * SEQ * DCAT;
  const unsigned short* kbase = kcat + bh * SEQ * DCAT;

  // ---- phase 1: scores = qcat . kcat (K=96 -> 3 WMMA steps), +mask bias ----
  {
    const int mt = wave >> 2;               // 0..1  (q row sub-tile)
    const int ntStart = (wave & 3) * 16;    // each wave: 16 key tiles
    const int m = lane & 15, half = lane >> 4;
    const int qrow = qt * 32 + mt * 16 + m;
    const unsigned short* ap = qbase + (size_t)qrow * DCAT + half * 8;

    FragU fa[3];
#pragma unroll
    for (int kf = 0; kf < 3; ++kf) {
      unpack8(*(const uint4*)(ap + kf * 32),      fa[kf].s);
      unpack8(*(const uint4*)(ap + kf * 32 + 16), fa[kf].s + 8);
    }
    for (int i = 0; i < 16; ++i) {
      const int nt  = ntStart + i;
      const int key = nt * 16 + m;
      const unsigned short* bp = kbase + (size_t)key * DCAT + half * 8;
      v8f c = {0.f, 0.f, 0.f, 0.f, 0.f, 0.f, 0.f, 0.f};
#pragma unroll
      for (int kf = 0; kf < 3; ++kf) {
        FragU fb;
        unpack8(*(const uint4*)(bp + kf * 32),     fb.s);
        unpack8(*(const uint4*)(bp + kf * 32 + 16), fb.s + 8);
        c = wmma_bf16(fa[kf], fb, c);
      }
      const float biasv = (1.0f - mask[(size_t)b * SEQ + key]) * NEGF;
#pragma unroll
      for (int j = 0; j < 8; ++j)
        sc[(mt * 16 + half * 8 + j) * 1024 + key] = c[j] + biasv;
    }
  }
  __syncthreads();

  // ---- phase 2: softmax over 1024 keys; fp32 probs -> global (NT),
  //               bf16 probs -> LDS for phase 3 ----
  {
    const int r = tid >> 3, seg = tid & 7;
    float* row = sc + r * 1024 + seg * 128;
    float mx = -3.0e38f;
#pragma unroll 4
    for (int i = 0; i < 128; ++i) mx = fmaxf(mx, row[i]);
    red[r * 8 + seg] = mx;
    __syncthreads();
    float rmax = red[r * 8];
#pragma unroll
    for (int i = 1; i < 8; ++i) rmax = fmaxf(rmax, red[r * 8 + i]);
    __syncthreads();
    float s = 0.f;
#pragma unroll 4
    for (int i = 0; i < 128; ++i) {
      const float e = __expf(row[i] - rmax);
      row[i] = e;
      s += e;
    }
    red[r * 8 + seg] = s;
    __syncthreads();
    float tot = 0.f;
#pragma unroll
    for (int i = 0; i < 8; ++i) tot += red[r * 8 + i];
    const float inv = 1.0f / tot;
    float* gout = probs + ((bh * SEQ) + (size_t)(qt * 32 + r)) * SEQ + seg * 128;
    unsigned short* prow = pb + r * 1024 + seg * 128;
#pragma unroll 8
    for (int i = 0; i < 128; i += 4) {
      float4 p = *(const float4*)(row + i);
      p.x *= inv; p.y *= inv; p.z *= inv; p.w *= inv;
      __builtin_nontemporal_store(__builtin_bit_cast(v4f, p),
                                  (v4f*)(gout + i));          // write-once stream
      uint2 pk;
      pk.x = (unsigned int)f2bf(p.x) | ((unsigned int)f2bf(p.y) << 16);
      pk.y = (unsigned int)f2bf(p.z) | ((unsigned int)f2bf(p.w) << 16);
      *(uint2*)(prow + i) = pk;
    }
  }
  __syncthreads();

  // ---- phase 3: ctx = probs @ v (64 cols), lctx = probs @ lv (16 cols) ----
  const unsigned short* vb  = vT  + bh * (size_t)(DHEAD * SEQ);
  const unsigned short* lvb = lvT + bh * (size_t)(DHL * SEQ);
  for (int t = wave; t < 10; t += 8) {
    const bool isl = (t >= 8);
    const int mt = isl ? (t - 8) : (t & 1);
    const int nt = isl ? 0 : (t >> 1);
    const int m = lane & 15, half = lane >> 4;
    const unsigned short* arow = pb + (mt * 16 + m) * 1024;
    const int col = nt * 16 + m;
    const unsigned short* bp = (isl ? lvb : vb) + (size_t)col * SEQ + half * 8;

    v8f c = {0.f, 0.f, 0.f, 0.f, 0.f, 0.f, 0.f, 0.f};
    for (int k = 0; k < SEQ; k += 32) {
      const int k0 = k + half * 8;
      FragU fa, fb;
      unpack8(*(const uint4*)(arow + k0),      fa.s);     // ds_load_b128
      unpack8(*(const uint4*)(arow + k0 + 16), fa.s + 8); // ds_load_b128
      unpack8(*(const uint4*)(bp + k),     fb.s);
      unpack8(*(const uint4*)(bp + k + 16), fb.s + 8);
      c = wmma_bf16(fa, fb, c);
    }
    const int srowBase = qt * 32 + mt * 16 + half * 8;
#pragma unroll
    for (int j = 0; j < 8; ++j) {
      const int s = srowBase + j;
      if (!isl)
        ctx[((size_t)b * SEQ + s) * DMODEL + h * DHEAD + col] = c[j];
      else
        lctx[((size_t)b * SEQ + s) * DLAY + h * DHL + col] = c[j];
    }
  }
}

// ---------------- host-side orchestration ----------------
extern "C" void kernel_launch(void* const* d_in, const int* in_sizes, int n_in,
                              void* d_out, int out_size, void* d_ws, size_t ws_size,
                              hipStream_t stream) {
  (void)in_sizes; (void)n_in; (void)out_size; (void)ws_size;
  const float* hs   = (const float*)d_in[0];
  const float* lhs  = (const float*)d_in[1];
  const float* mask = (const float*)d_in[2];
  const float* Wq  = (const float*)d_in[3];  const float* bq  = (const float*)d_in[4];
  const float* Wk  = (const float*)d_in[5];  const float* bk  = (const float*)d_in[6];
  const float* Wv  = (const float*)d_in[7];  const float* bv  = (const float*)d_in[8];
  const float* Wlq = (const float*)d_in[9];  const float* blq = (const float*)d_in[10];
  const float* Wlk = (const float*)d_in[11]; const float* blk = (const float*)d_in[12];
  const float* Wlv = (const float*)d_in[13]; const float* blv = (const float*)d_in[14];

  // output partitioning (fp32): ctx | lctx | probs
  float* ctx   = (float*)d_out;
  float* lctx  = ctx + (size_t)BATCH * SEQ * DMODEL;
  float* probs = lctx + (size_t)BATCH * SEQ * DLAY;

  // workspace carve-up (bf16 stored as unsigned short)
  const size_t nX  = (size_t)BATCH * SEQ * DMODEL;     // 6291456
  const size_t nXL = (size_t)BATCH * SEQ * DLAY;       // 1572864
  const size_t nW  = (size_t)DMODEL * DMODEL;          // 589824
  const size_t nWl = (size_t)DLAY * DLAY;              // 36864
  const size_t nCat = (size_t)BATCH * HEADS * SEQ * DCAT;   // 9437184
  const size_t nVT  = (size_t)BATCH * HEADS * DHEAD * SEQ;  // 6291456
  const size_t nLVT = (size_t)BATCH * HEADS * DHL * SEQ;    // 1572864

  char* ws = (char*)d_ws;
  size_t off = 0;
  auto carve = [&](size_t elems) {
    unsigned short* p = (unsigned short*)(ws + off);
    off = (off + elems * 2 + 255) & ~(size_t)255;
    return p;
  };
  unsigned short* X16   = carve(nX);
  unsigned short* XL16  = carve(nXL);
  unsigned short* Wq16  = carve(nW);
  unsigned short* Wk16  = carve(nW);
  unsigned short* Wv16  = carve(nW);
  unsigned short* Wlq16 = carve(nWl);
  unsigned short* Wlk16 = carve(nWl);
  unsigned short* Wlv16 = carve(nWl);
  unsigned short* qcat  = carve(nCat);
  unsigned short* kcat  = carve(nCat);
  unsigned short* vT    = carve(nVT);
  unsigned short* lvT   = carve(nLVT);

  // 1) fp32 -> bf16 conversions
  auto cvt = [&](const float* s, unsigned short* d, size_t n) {
    cvt_f32_bf16<<<dim3((unsigned)((n + 255) / 256)), dim3(256), 0, stream>>>(s, d, (int)n);
  };
  cvt(hs, X16, nX);     cvt(lhs, XL16, nXL);
  cvt(Wq, Wq16, nW);    cvt(Wk, Wk16, nW);    cvt(Wv, Wv16, nW);
  cvt(Wlq, Wlq16, nWl); cvt(Wlk, Wlk16, nWl); cvt(Wlv, Wlv16, nWl);

  // 2) zero qcat/kcat (zero pad columns 80..95 must be 0)
  (void)hipMemsetAsync(qcat, 0, nCat * 2, stream);
  (void)hipMemsetAsync(kcat, 0, nCat * 2, stream);

  // 3) projections (scale folds 1/sqrt(DH) into q, 1/sqrt(DHL) into lq)
  const int M = BATCH * SEQ;  // 8192
  dim3 tpb(128);
  dim3 gD(M / 16, DMODEL / 64);  // (512, 12)
  dim3 gL(M / 16, DLAY / 64);    // (512, 3)
  gemm_proj_wmma<<<gD, tpb, 0, stream>>>(X16,  Wq16,  bq,  qcat, DMODEL, DMODEL, 0.125f, 2, DHEAD, 0);
  gemm_proj_wmma<<<gD, tpb, 0, stream>>>(X16,  Wk16,  bk,  kcat, DMODEL, DMODEL, 1.0f,   2, DHEAD, 0);
  gemm_proj_wmma<<<gL, tpb, 0, stream>>>(XL16, Wlq16, blq, qcat, DLAY,   DLAY,   0.25f,  2, DHL,   DHEAD);
  gemm_proj_wmma<<<gL, tpb, 0, stream>>>(XL16, Wlk16, blk, kcat, DLAY,   DLAY,   1.0f,   2, DHL,   DHEAD);
  gemm_proj_wmma<<<gD, tpb, 0, stream>>>(X16,  Wv16,  bv,  vT,   DMODEL, DMODEL, 1.0f,   1, 0, 0);
  gemm_proj_wmma<<<gL, tpb, 0, stream>>>(XL16, Wlv16, blv, lvT,  DLAY,   DLAY,   1.0f,   1, 0, 0);

  // 4) fused attention
  const int smemBytes = 32 * 1024 * 4 + 32 * 8 * 4 + 32 * 1024 * 2; // 197632 B
  (void)hipFuncSetAttribute(reinterpret_cast<const void*>(attn_wmma),
                            hipFuncAttributeMaxDynamicSharedMemorySize, smemBytes);
  attn_wmma<<<dim3(BATCH * HEADS * (SEQ / 32)), dim3(256), smemBytes, stream>>>(
      qcat, kcat, vT, lvT, mask, ctx, lctx, probs);
}
